// DualWarping_22771916603583
// MI455X (gfx1250) — matrix operand order repaired and checked by reading, not compiled, not run on hardware
//
#include <hip/hip_runtime.h>
#include <hip/hip_bf16.h>
#include <cstdint>

typedef __bf16 bf16_t;
typedef __bf16 v16bf __attribute__((ext_vector_type(16)));
typedef float  v8f   __attribute__((ext_vector_type(8)));

#define NB   4
#define CF   96
#define CIN1 99
#define HF   180
#define WF   320
#define HWF  (HF*WF)      // 57600
#define HI   720
#define WI   1280
#define HWI  (HI*WI)      // 921600

#define TM   128          // pixels per workgroup (M tile)
#define KC   32           // K chunk (= WMMA bf16 K)
#define APAD 40           // bf16 row pad: 80B rows -> 16B aligned, bank-spread
#define CPAD 132          // f32 epilogue tile pad

#define KPAD1 896         // ceil(99*9 / 32) * 32
#define KPAD2 864         // 96*9 (already multiple of 32)

__device__ __forceinline__ float clampf(float v, float lo, float hi) {
    return fminf(fmaxf(v, lo), hi);
}

// ---------------------------------------------------------------------------
// Kernel A: full-res backward warp of img with border padding + valid mask.
// ---------------------------------------------------------------------------
__global__ void warp_img_kernel(const float* __restrict__ img,
                                const float* __restrict__ flow,
                                float* __restrict__ out1) {
    int idx = blockIdx.x * blockDim.x + threadIdx.x;
    if (idx >= NB * HWI) return;
    int b = idx / HWI;
    int p = idx - b * HWI;
    int y = p / WI, x = p - (p / WI) * WI;

    const float* fb = flow + (size_t)b * 2 * HWI;
    float gx = (float)x + fb[p];
    float gy = (float)y + fb[p + HWI];
    float valid = (gx >= 0.f && gx <= (float)(WI - 1) &&
                   gy >= 0.f && gy <= (float)(HI - 1)) ? 1.f : 0.f;
    float xs = clampf(gx, 0.f, (float)(WI - 1));
    float ys = clampf(gy, 0.f, (float)(HI - 1));
    float x0f = floorf(xs), y0f = floorf(ys);
    float wx = xs - x0f, wy = ys - y0f;
    int x0 = (int)x0f, y0 = (int)y0f;
    int x1 = min(x0 + 1, WI - 1), y1 = min(y0 + 1, HI - 1);
    float w00 = (1.f - wx) * (1.f - wy), w01 = wx * (1.f - wy);
    float w10 = (1.f - wx) * wy,         w11 = wx * wy;

    #pragma unroll
    for (int c = 0; c < 3; ++c) {
        const float* ip = img + (size_t)(b * 3 + c) * HWI;
        float v = ip[y0 * WI + x0] * w00 + ip[y0 * WI + x1] * w01 +
                  ip[y1 * WI + x0] * w10 + ip[y1 * WI + x1] * w11;
        out1[(size_t)(b * 3 + c) * HWI + p] = v * valid;
    }
}

// ---------------------------------------------------------------------------
// Kernel B: fused flow-downsample (align_corners=False, *0.25) + feature warp.
// Writes f32 residual copy and bf16 cat channels [0,96).
// ---------------------------------------------------------------------------
__global__ void warp_feat_kernel(const float* __restrict__ feat,
                                 const float* __restrict__ flow,
                                 float* __restrict__ wf32,
                                 bf16_t* __restrict__ cat) {
    int idx = blockIdx.x * blockDim.x + threadIdx.x;
    if (idx >= NB * HWF) return;
    int b = idx / HWF;
    int p = idx - b * HWF;
    int yf = p / WF, xf = p - (p / WF) * WF;

    float yc = clampf(((float)yf + 0.5f) * 4.f - 0.5f, 0.f, (float)(HI - 1));
    float xc = clampf(((float)xf + 0.5f) * 4.f - 0.5f, 0.f, (float)(WI - 1));
    int y0 = (int)floorf(yc), x0 = (int)floorf(xc);
    int y1 = min(y0 + 1, HI - 1), x1 = min(x0 + 1, WI - 1);
    float wy = yc - (float)y0, wx = xc - (float)x0;
    float s00 = (1.f - wx) * (1.f - wy), s01 = wx * (1.f - wy);
    float s10 = (1.f - wx) * wy,         s11 = wx * wy;
    const float* f0 = flow + (size_t)b * 2 * HWI;
    const float* f1 = f0 + HWI;
    float fs0 = (f0[y0 * WI + x0] * s00 + f0[y0 * WI + x1] * s01 +
                 f0[y1 * WI + x0] * s10 + f0[y1 * WI + x1] * s11) * 0.25f;
    float fs1 = (f1[y0 * WI + x0] * s00 + f1[y0 * WI + x1] * s01 +
                 f1[y1 * WI + x0] * s10 + f1[y1 * WI + x1] * s11) * 0.25f;

    float gx = (float)xf + fs0, gy = (float)yf + fs1;
    float valid = (gx >= 0.f && gx <= (float)(WF - 1) &&
                   gy >= 0.f && gy <= (float)(HF - 1)) ? 1.f : 0.f;
    float xs = clampf(gx, 0.f, (float)(WF - 1));
    float ys = clampf(gy, 0.f, (float)(HF - 1));
    int fx0 = (int)floorf(xs), fy0 = (int)floorf(ys);
    int fx1 = min(fx0 + 1, WF - 1), fy1 = min(fy0 + 1, HF - 1);
    float awx = xs - (float)fx0, awy = ys - (float)fy0;
    float w00 = (1.f - awx) * (1.f - awy), w01 = awx * (1.f - awy);
    float w10 = (1.f - awx) * awy,         w11 = awx * awy;
    int o00 = fy0 * WF + fx0, o01 = fy0 * WF + fx1;
    int o10 = fy1 * WF + fx0, o11 = fy1 * WF + fx1;

    for (int c = 0; c < CF; ++c) {
        const float* ip = feat + (size_t)(b * CF + c) * HWF;
        float v = (ip[o00] * w00 + ip[o01] * w01 + ip[o10] * w10 + ip[o11] * w11) * valid;
        wf32[(size_t)(b * CF + c) * HWF + p] = v;
        cat[(size_t)(b * CIN1 + c) * HWF + p] = (bf16_t)v;
    }
}

// ---------------------------------------------------------------------------
// Kernel C: downsample warped_img (align_corners=False) -> bf16 cat ch [96,99).
// ---------------------------------------------------------------------------
__global__ void downsample_img_kernel(const float* __restrict__ wimg,
                                      bf16_t* __restrict__ cat) {
    int idx = blockIdx.x * blockDim.x + threadIdx.x;
    if (idx >= NB * 3 * HWF) return;
    int b = idx / (3 * HWF);
    int r = idx - b * (3 * HWF);
    int c = r / HWF;
    int p = r - c * HWF;
    int yf = p / WF, xf = p - (p / WF) * WF;

    float yc = clampf(((float)yf + 0.5f) * 4.f - 0.5f, 0.f, (float)(HI - 1));
    float xc = clampf(((float)xf + 0.5f) * 4.f - 0.5f, 0.f, (float)(WI - 1));
    int y0 = (int)floorf(yc), x0 = (int)floorf(xc);
    int y1 = min(y0 + 1, HI - 1), x1 = min(x0 + 1, WI - 1);
    float wy = yc - (float)y0, wx = xc - (float)x0;
    const float* ip = wimg + (size_t)(b * 3 + c) * HWI;
    float v = ip[y0 * WI + x0] * (1.f - wx) * (1.f - wy) +
              ip[y0 * WI + x1] * wx * (1.f - wy) +
              ip[y1 * WI + x0] * (1.f - wx) * wy +
              ip[y1 * WI + x1] * wx * wy;
    cat[(size_t)(b * CIN1 + CF + c) * HWF + p] = (bf16_t)v;
}

// ---------------------------------------------------------------------------
// Kernel W: pack OIHW f32 weights into n-major bf16 [n][Kpad], k = e*Cin + c
// (spatial tap outer, channel inner), zero-padded so the GEMM needs no K tail.
// ---------------------------------------------------------------------------
__global__ void pack_weights_kernel(const float* __restrict__ W1,
                                    const float* __restrict__ W2,
                                    bf16_t* __restrict__ wBt1,
                                    bf16_t* __restrict__ wBt2) {
    int idx = blockIdx.x * blockDim.x + threadIdx.x;
    if (idx < 96 * KPAD1) {
        int n = idx / KPAD1, k = idx - n * KPAD1;
        float v = 0.f;
        if (k < CIN1 * 9) {
            int e = k / CIN1, c = k - e * CIN1;
            v = W1[(size_t)n * (CIN1 * 9) + c * 9 + e];
        }
        wBt1[(size_t)n * KPAD1 + k] = (bf16_t)v;
    }
    if (idx < 96 * KPAD2) {
        int n = idx / KPAD2, k = idx - n * KPAD2;
        int e = k / CF, c = k - e * CF;
        wBt2[(size_t)n * KPAD2 + k] = (bf16_t)W2[(size_t)n * (CF * 9) + c * 9 + e];
    }
}

// ---------------------------------------------------------------------------
// Kernel D: implicit-GEMM 3x3 conv via v_wmma_f32_16x16x32_bf16.
// K order: k = e*Cin + c  (e = (ky+1)*3 + (kx+1), c = input channel).
// A staged in double-buffered LDS with a division-free incremental walker;
// B fragments read straight from the packed [n][Kpad] global buffer (L2/WGP$
// resident, two b128 per fragment). Software-pipelined: prefetch chunk k+1 to
// registers, run 6 WMMAs on chunk k, store, one barrier per chunk.
// ---------------------------------------------------------------------------
__launch_bounds__(256)
__global__ void conv_gemm_kernel(const bf16_t* __restrict__ in, int Cin,
                                 const bf16_t* __restrict__ wBt, int Kpad,
                                 const float* __restrict__ bias,
                                 const float* __restrict__ residual,
                                 bf16_t* __restrict__ out_bf,
                                 float* __restrict__ out_f,
                                 int mode) {
    __shared__ __align__(16) bf16_t As[2][TM][APAD];  // [buf][m][k] bf16
    __shared__ float Cs[CF][CPAD];                    // epilogue transpose tile

    const int tid = threadIdx.x;
    const int tiles_per_b = HWF / TM;                 // 450
    const int b  = blockIdx.x / tiles_per_b;
    const int m0 = (blockIdx.x - b * tiles_per_b) * TM;
    const int lane = tid & 31;
    const int wave = tid >> 5;
    const int h16  = lane >> 4;
    const int l16  = lane & 15;

    const int t7 = tid >> 7;                          // which 16-k half this thread stages
    const int ml = tid & 127;                         // staged pixel (fixed per thread)

    const bf16_t* __restrict__ inb = in + (size_t)b * Cin * HWF;
    const int m = m0 + ml;
    const int y = m / WF;
    const int x = m - y * WF;
    const bf16_t* __restrict__ pbase = inb + m;

    // 9-bit halo validity mask over taps e=0..8; (vmask>>e)&1 == 0 for e>=9
    // also kills the padded K tail automatically.
    int vmask = 0;
    #pragma unroll
    for (int e = 0; e < 9; ++e) {
        int ky = e / 3 - 1, kx = e % 3 - 1;           // compile-time constants
        int yy = y + ky, xx = x + kx;
        if (yy >= 0 && yy < HF && xx >= 0 && xx < WF) vmask |= (1 << e);
    }

    // incremental k -> (c, e) walker; off = ky*WF + kx, coff = c*HWF
    int wc = t7 * 16;
    int we = 0, we3 = 0;
    int woff = -WF - 1;
    int wcoff = wc * HWF;

    union AVec { bf16_t h[16]; uint4 u[2]; v16bf v; };

    v8f acc[6];
    #pragma unroll
    for (int i = 0; i < 6; ++i) acc[i] = (v8f){0.f,0.f,0.f,0.f,0.f,0.f,0.f,0.f};

    const int numChunks = Kpad / KC;                  // 28 or 27, no tail
    AVec va;

    // ---- stage loader: 16 consecutive k, then skip the other half (16 more)
    auto stage_load = [&]() {
        #pragma unroll
        for (int i = 0; i < 16; ++i) {
            bool valid = ((vmask >> we) & 1) != 0;
            va.h[i] = valid ? pbase[wcoff + woff] : (bf16_t)0.f;
            ++wc; wcoff += HWF;
            if (wc >= Cin) {
                wc = 0; wcoff = 0;
                ++we;
                if (++we3 == 3) { we3 = 0; woff += WF - 2; } else { ++woff; }
            }
        }
        wc += 16; wcoff += 16 * HWF;
        if (wc >= Cin) {
            wc -= Cin; wcoff -= Cin * HWF;
            ++we;
            if (++we3 == 3) { we3 = 0; woff += WF - 2; } else { ++woff; }
        }
    };

    // ---- prologue: stage chunk 0 into buffer 0
    stage_load();
    {
        uint4* dst = reinterpret_cast<uint4*>(&As[0][ml][t7 * 16]);
        dst[0] = va.u[0];
        dst[1] = va.u[1];
    }
    __syncthreads();

    for (int kc = 0; kc < numChunks; ++kc) {
        const int cur = kc & 1;
        const bool more = (kc + 1) < numChunks;
        const int k0 = kc * KC;

        if (more) stage_load();                       // global loads for chunk kc+1

        // ---- B fragments: two contiguous b128 from packed global weights
        AVec bfrag[6];
        #pragma unroll
        for (int nt = 0; nt < 6; ++nt) {
            const uint4* pb = reinterpret_cast<const uint4*>(
                wBt + (size_t)(nt * 16 + l16) * Kpad + k0 + h16 * 16);
            bfrag[nt].u[0] = pb[0];
            bfrag[nt].u[1] = pb[1];
        }
        // ---- A fragment: two contiguous b128 from LDS
        AVec af;
        {
            const uint4* pa = reinterpret_cast<const uint4*>(&As[cur][wave * 16 + l16][0]);
            af.u[0] = pa[2 * h16];
            af.u[1] = pa[2 * h16 + 1];
        }
        #pragma unroll
        for (int nt = 0; nt < 6; ++nt) {
            acc[nt] = __builtin_amdgcn_wmma_f32_16x16x32_bf16(
                false, af.v, false, bfrag[nt].v, (short)0, acc[nt], false, false);
        }

        if (more) {
            uint4* dst = reinterpret_cast<uint4*>(&As[cur ^ 1][ml][t7 * 16]);
            dst[0] = va.u[0];
            dst[1] = va.u[1];
        }
        __syncthreads();
    }

    // ---- epilogue: transpose through LDS for channel-row-contiguous stores
    // D layout: VGPR v -> M = v + 8*h16 (within wave's 16-row slice), N = l16.
    #pragma unroll
    for (int nt = 0; nt < 6; ++nt) {
        int n = nt * 16 + l16;
        #pragma unroll
        for (int v = 0; v < 8; ++v) {
            Cs[n][wave * 16 + v + 8 * h16] = acc[nt][v];
        }
    }
    __syncthreads();

    if (mode == 0) {
        #pragma unroll 4
        for (int i = 0; i < 48; ++i) {
            int idx = i * 256 + tid;
            int n  = idx >> 7;
            int ml2 = idx & 127;
            float v = fmaxf(Cs[n][ml2] + bias[n], 0.f);
            out_bf[(size_t)(b * CF + n) * HWF + m0 + ml2] = (bf16_t)v;
        }
    } else {
        #pragma unroll 4
        for (int i = 0; i < 48; ++i) {
            int idx = i * 256 + tid;
            int n  = idx >> 7;
            int ml2 = idx & 127;
            size_t off = (size_t)(b * CF + n) * HWF + m0 + ml2;
            out_f[off] = Cs[n][ml2] + bias[n] + residual[off];
        }
    }
}

// ---------------------------------------------------------------------------
extern "C" void kernel_launch(void* const* d_in, const int* in_sizes, int n_in,
                              void* d_out, int out_size, void* d_ws, size_t ws_size,
                              hipStream_t stream) {
    (void)in_sizes; (void)n_in; (void)out_size; (void)ws_size;
    const float* img  = (const float*)d_in[0];
    const float* feat = (const float*)d_in[1];
    const float* flow = (const float*)d_in[2];
    const float* W1   = (const float*)d_in[3];
    const float* b1   = (const float*)d_in[4];
    const float* W2   = (const float*)d_in[5];
    const float* b2   = (const float*)d_in[6];

    float* out0 = (float*)d_out;                          // warped_feat + fused
    float* out1 = out0 + (size_t)NB * CF * HWF;           // warped_img

    // workspace layout (bytes)
    char* ws = (char*)d_ws;
    float*  wf32 = (float*)(ws);                                       //  88,473,600 B
    bf16_t* cat  = (bf16_t*)(ws + 88473600u);                          //  45,619,200 B
    bf16_t* h    = (bf16_t*)(ws + 88473600u + 45619200u);              //  44,236,800 B
    bf16_t* wBt1 = (bf16_t*)(ws + 88473600u + 45619200u + 44236800u);  //     172,032 B
    bf16_t* wBt2 = (bf16_t*)(ws + 88473600u + 45619200u + 44236800u + 172032u);

    warp_img_kernel<<<(NB * HWI + 255) / 256, 256, 0, stream>>>(img, flow, out1);
    warp_feat_kernel<<<(NB * HWF + 255) / 256, 256, 0, stream>>>(feat, flow, wf32, cat);
    downsample_img_kernel<<<(NB * 3 * HWF + 255) / 256, 256, 0, stream>>>(out1, cat);
    pack_weights_kernel<<<(96 * KPAD1 + 255) / 256, 256, 0, stream>>>(W1, W2, wBt1, wBt2);

    const int gemm_blocks = NB * (HWF / TM);   // 1800
    conv_gemm_kernel<<<gemm_blocks, 256, 0, stream>>>(cat, CIN1, wBt1, KPAD1, b1,
                                                      nullptr, h, nullptr, 0);
    conv_gemm_kernel<<<gemm_blocks, 256, 0, stream>>>(h, CF, wBt2, KPAD2, b2,
                                                      wf32, nullptr, out0, 1);
}
